// ModulatedConv2dStyleTensor_88210038325478
// MI455X (gfx1250) — matrix operand order repaired and checked by reading, not compiled, not run on hardware
//
#include <hip/hip_runtime.h>

typedef __attribute__((ext_vector_type(16))) _Float16 v16h;
typedef __attribute__((ext_vector_type(8)))  _Float16 v8h;
typedef __attribute__((ext_vector_type(8)))  float    v8f;

typedef unsigned int u32;
typedef unsigned long long u64;
typedef u32 u32x4 __attribute__((ext_vector_type(4)));
typedef int i32x4 __attribute__((ext_vector_type(4)));
typedef int i32x8 __attribute__((ext_vector_type(8)));

#define B_N   8
#define C_IN  512
#define C_OUT 512
#define H_    64
#define W_    64
#define NPOS  4096
#define SD_   512
#define NS_   24
#define SDNS  536
#define HP    66          // padded spatial extent
#define NPP   (HP*HP)     // 4356 padded positions
#define XST   40          // LDS row stride in halfs: 32 data + 8 pad (TDM pad 16DW+4DW)

// ---------------- TDM: load a (up to 3D) f16 tile into LDS -------------------
// dim0 (x): contiguous in memory, tile0 elements (row = 32 halfs = 16 DWORDs),
// hardware appends 4 DWORDs of LDS padding after each row -> stride XST halfs.
__device__ __forceinline__
void tdm_load_f16(u32 ldsAddr, const void* gptr,
                  u32 td0, u32 td1, u32 tile0, u32 tile1, u32 tile2,
                  u32 stride0, u64 stride1, u32 td2) {
  const u64 ga = (u64)(size_t)gptr;
  u32x4 g0;
  g0[0] = 1u;                                   // count=1 (valid), user mode
  g0[1] = ldsAddr;                              // LDS byte address
  g0[2] = (u32)ga;                              // global_addr[31:0]
  g0[3] = (u32)(ga >> 32) | (2u << 30);         // global_addr[56:32] | type=2
  i32x8 g1;
  g1[0] = (int)((1u << 16) |                    // data_size = 2 bytes
                (1u << 20) |                    // pad_enable
                (3u << 22) |                    // pad_interval: 16 DWORDs
                (3u << 25));                    // pad_amount: 4 DWORDs
  g1[1] = (int)(td0 << 16);                     // tensor_dim0[15:0]
  g1[2] = (int)((td0 >> 16) | (td1 << 16));     // tensor_dim0[31:16] | tensor_dim1[15:0]
  g1[3] = (int)((td1 >> 16) | (tile0 << 16));   // tensor_dim1[31:16] | tile_dim0
  g1[4] = (int)(tile1 | (tile2 << 16));         // tile_dim1 | tile_dim2
  g1[5] = (int)stride0;                         // tensor_dim0_stride[31:0]
  g1[6] = (int)((u32)(stride1 & 0xFFFFu) << 16);// dim0_stride[47:32]=0 | dim1_stride[15:0]
  g1[7] = (int)(u32)(stride1 >> 16);            // tensor_dim1_stride[47:16]
  i32x4 g2;
  g2[0] = (int)td2;                             // tensor_dim2
  g2[1] = 0; g2[2] = 0; g2[3] = 0;              // no iterate / dim3
  i32x4 g3 = {0, 0, 0, 0};
  i32x8 g4 = {0, 0, 0, 0, 0, 0, 0, 0};          // extra group (clang-23 6-arg form)
  __builtin_amdgcn_tensor_load_to_lds(g0, g1, g2, g3, g4, 0);
}

// ---- kernel 1: mean over H*W of each segm channel (24 channels) -------------
__global__ __launch_bounds__(256)
void k_segmean(const float* __restrict__ segm, float* __restrict__ segmmean) {
  __shared__ float red[256];
  const int c = blockIdx.x;
  float s = 0.f;
  for (int p = threadIdx.x; p < NPOS; p += 256) s += segm[c * NPOS + p];
  red[threadIdx.x] = s;
  __syncthreads();
  for (int w = 128; w > 0; w >>= 1) {
    if (threadIdx.x < w) red[threadIdx.x] += red[threadIdx.x + w];
    __syncthreads();
  }
  if (threadIdx.x == 0) segmmean[c] = red[0] * (1.0f / NPOS);
}

// ---- kernel 2: styleterm[b,o] (incl. bias) and style_mean[b,o] --------------
__global__ __launch_bounds__(256)
void k_style(const float* __restrict__ style, const float* __restrict__ scw,
             const float* __restrict__ scb, const float* __restrict__ segmmean,
             float* __restrict__ styleterm, float* __restrict__ stylemean) {
  const int gid = blockIdx.x * 256 + threadIdx.x;   // 8*512
  const int b = gid >> 9, o = gid & 511;
  const float sc = rsqrtf((float)SDNS);
  float acc = 0.f;
  for (int c = 0; c < SD_; ++c) acc += style[b * SD_ + c] * scw[o * SDNS + c];
  const float st = sc * acc + scb[o];
  float acc2 = 0.f;
  #pragma unroll
  for (int c = 0; c < NS_; ++c) acc2 += scw[o * SDNS + SD_ + c] * segmmean[c];
  styleterm[gid] = st;
  stylemean[gid] = st + sc * acc2;
}

// ---- kernel 3: demod[b,cout] ------------------------------------------------
__global__ __launch_bounds__(256)
void k_demod(const float* __restrict__ weight, const float* __restrict__ stylemean,
             float* __restrict__ demod) {
  const int gid = blockIdx.x * 256 + threadIdx.x;   // 8*512
  const int b = gid >> 9, co = gid & 511;
  float acc = 0.f;
  for (int ci = 0; ci < C_IN; ++ci) {
    const float* wp = weight + co * (C_IN * 9) + ci * 9;
    float wsq = 0.f;
    #pragma unroll
    for (int j = 0; j < 9; ++j) wsq += wp[j] * wp[j];
    const float smv = stylemean[b * C_IN + ci];
    acc += wsq * smv * smv;
  }
  demod[gid] = rsqrtf(acc * (1.0f / 4608.0f) + 1e-8f);
}

// ---- kernel 4: repack weight fp32 [cout][cin][kh][kw] -> f16 [tap][cout][cin]
__global__ __launch_bounds__(256)
void k_wcvt(const float* __restrict__ weight, _Float16* __restrict__ w16) {
  const int gid = blockIdx.x * 256 + threadIdx.x;
  if (gid >= 9 * C_OUT * C_IN) return;
  const int tap = gid / (C_OUT * C_IN);
  const int rem = gid - tap * (C_OUT * C_IN);
  const int co = rem >> 9, ci = rem & 511;
  w16[gid] = (_Float16)weight[co * (C_IN * 9) + ci * 9 + tap];
}

// ---- kernel 5: modulated input, zero-padded, channel-last -------------------
// xpad[b][hp*66+wp][cin] f16 : hp,wp in [0,66), borders zero, interior = x*s.
// grid: (4356 padded positions, 2 cin chunks), 256 threads (cin fastest -> coalesced stores)
__global__ __launch_bounds__(256)
void k_xmod(const float* __restrict__ input, const float* __restrict__ scw,
            const float* __restrict__ segm, const float* __restrict__ styleterm,
            _Float16* __restrict__ xpad) {
  const int pp = blockIdx.x;                          // 0..4355
  const int ci = blockIdx.y * 256 + threadIdx.x;      // 0..511
  const int hp = pp / HP, wp = pp - hp * HP;
  float vals[B_N];
  #pragma unroll
  for (int b = 0; b < B_N; ++b) vals[b] = 0.f;
  if (hp >= 1 && hp <= 64 && wp >= 1 && wp <= 64) {
    const int p = (hp - 1) * W_ + (wp - 1);
    const float sc = rsqrtf((float)SDNS);
    float segt = 0.f;
    #pragma unroll
    for (int c = 0; c < NS_; ++c)
      segt += scw[ci * SDNS + SD_ + c] * segm[c * NPOS + p];
    segt *= sc;
    #pragma unroll
    for (int b = 0; b < B_N; ++b) {
      const float s = styleterm[b * C_IN + ci] + segt;
      vals[b] = input[(b * C_IN + ci) * NPOS + p] * s;
    }
  }
  #pragma unroll
  for (int b = 0; b < B_N; ++b)
    xpad[(b * NPP + pp) * C_IN + ci] = (_Float16)vals[b];
}

// ---- kernel 6: implicit-GEMM 3x3 conv: TDM tiles + WMMA f16->f32 ------------
// grid: x = 32 pos tiles (128 pos = 2 image rows), y = 4 cout tiles, z = batch
__global__ __launch_bounds__(256)
void k_conv_wmma(const _Float16* __restrict__ xpad, const _Float16* __restrict__ w16,
                 const float* __restrict__ demod, float* __restrict__ out) {
  __shared__ __align__(16) _Float16 Xs[264 * XST];      // [r*66+c][cin] via TDM (padded rows)
  __shared__ __align__(16) _Float16 As[2 * 128 * XST];  // double-buffered [cout][cin]
  const int posBase  = blockIdx.x * 128;
  const int coutBase = blockIdx.y * 128;
  const int b        = blockIdx.z;
  const int t    = threadIdx.x;
  const int lane = t & 31;
  const int wid  = t >> 5;          // 8 waves: 4 (M) x 2 (N)
  const int waveM = wid >> 1;
  const int waveN = wid & 1;
  const int lm  = lane & 15;
  const int lhi = lane >> 4;
  const int koff  = lhi * 8;        // A-fragment K offset (ISA 16-bit A layout)
  const int kbase = lhi * 16;       // added to br, cancelled in the B loads (B uses koff)
  const int h0 = posBase >> 6;      // first image row of this 2-row tile

  const u32 xsLds = (u32)(size_t)&Xs[0];     // low 32 bits of flat LDS addr = LDS offset
  const u32 asLds0 = (u32)(size_t)&As[0];
  const u32 asLds1 = (u32)(size_t)&As[128 * XST];

  v8f acc[2][4];
  v8f zero = {};
  #pragma unroll
  for (int i = 0; i < 2; ++i)
    #pragma unroll
    for (int j = 0; j < 4; ++j) acc[i][j] = zero;

  for (int cc = 0; cc < 16; ++cc) {          // cin chunks of 32
    const int cin0 = cc * 32;
    __syncthreads();                          // all reads of Xs/As finished
    if (wid == 0) {
      // B tile: 3D (cin x 66 cols x 4 rows) from the pre-padded channel-last buffer.
      tdm_load_f16(xsLds, xpad + ((size_t)(b * NPP + h0 * HP)) * C_IN + cin0,
                   /*td0*/32, /*td1*/HP, /*tile0*/32, /*tile1*/HP, /*tile2*/4,
                   /*stride0*/C_IN, /*stride1*/(u64)HP * C_IN, /*td2*/4);
      // A tile for tap 0: 2D (cin x 128 couts)
      tdm_load_f16(asLds0, w16 + 0 * (C_OUT * C_IN) + coutBase * C_IN + cin0,
                   32, 128, 32, 128, 0, C_IN, 0, 0);
    }
    __builtin_amdgcn_s_wait_tensorcnt(0);
    __syncthreads();                          // Xs + As[0] visible

    #pragma unroll 1
    for (int tap = 0; tap < 9; ++tap) {
      const int buf = tap & 1;
      const int kh = tap / 3, kw = tap - kh * 3;
      if (tap < 8 && wid == 0) {              // prefetch next tap's weights via TDM
        tdm_load_f16(buf ? asLds0 : asLds1,
                     w16 + (tap + 1) * (C_OUT * C_IN) + coutBase * C_IN + cin0,
                     32, 128, 32, 128, 0, C_IN, 0, 0);
      }
      const _Float16* Ab = As + buf * (128 * XST);

      v16h af[2];
      #pragma unroll
      for (int mf = 0; mf < 2; ++mf) {
        const _Float16* ar = Ab + (waveM * 32 + mf * 16 + lm) * XST;
        ((v8h*)&af[mf])[0] = *(const v8h*)(ar + koff);        // K koff..koff+7
        ((v8h*)&af[mf])[1] = *(const v8h*)(ar + 16 + koff);   // K 16+koff..
      }
      #pragma unroll
      for (int nf = 0; nf < 4; ++nf) {
        // tap shift is pure LDS addressing into the extended tile
        const _Float16* br =
            Xs + ((waveN + kh) * HP + nf * 16 + lm + kw) * XST + kbase;
        v16h bf;
        ((v8h*)&bf)[0] = *(const v8h*)(br - kbase + koff);       // K koff..koff+7
        ((v8h*)&bf)[1] = *(const v8h*)(br - kbase + 16 + koff);  // K 16+koff..16+koff+7
        #pragma unroll
        for (int mf = 0; mf < 2; ++mf) {
          acc[mf][nf] = __builtin_amdgcn_wmma_f32_16x16x32_f16(
              false, af[mf], false, bf, (short)0, acc[mf][nf], false, false);
        }
      }
      if (tap < 8) {
        __builtin_amdgcn_s_wait_tensorcnt(0);   // next A tile landed (overlapped DMA)
        __syncthreads();
      }
    }
  }

  // Store with factored-out scaling: scale^2 / sqrt(H*W) * demod[b,cout]
  const float kFinal = 1.0f / (4608.0f * 64.0f);
  #pragma unroll
  for (int mf = 0; mf < 2; ++mf) {
    #pragma unroll
    for (int v = 0; v < 8; ++v) {
      const int co = coutBase + waveM * 32 + mf * 16 + lhi * 8 + v; // C layout: M=v(+8*lhi)
      const float fs = demod[b * C_OUT + co] * kFinal;
      #pragma unroll
      for (int nf = 0; nf < 4; ++nf) {
        const int pos = posBase + waveN * 64 + nf * 16 + lm;        // C layout: N=lane%16
        out[((b * C_OUT + co) * NPOS) + pos] = acc[mf][nf][v] * fs;
      }
    }
  }
}

extern "C" void kernel_launch(void* const* d_in, const int* in_sizes, int n_in,
                              void* d_out, int out_size, void* d_ws, size_t ws_size,
                              hipStream_t stream) {
  const float* input  = (const float*)d_in[0];
  const float* style  = (const float*)d_in[1];
  const float* scw    = (const float*)d_in[2];
  const float* scb    = (const float*)d_in[3];
  const float* weight = (const float*)d_in[4];
  const float* segm   = (const float*)d_in[5];
  float* out = (float*)d_out;

  char* ws = (char*)d_ws;
  float*    segmmean  = (float*)(ws);                       // 24 f
  float*    styleterm = (float*)(ws + 256);                 // 8*512 f
  float*    stylemean = (float*)(ws + 256 + 16384);         // 8*512 f
  float*    demod     = (float*)(ws + 256 + 2 * 16384);     // 8*512 f
  _Float16* w16       = (_Float16*)(ws + 256 + 3 * 16384);  // 9*512*512 h = 4.5 MiB
  _Float16* xpad      = (_Float16*)(ws + 256 + 3 * 16384 + 9 * 512 * 512 * 2); // 34 MiB

  k_segmean<<<24, 256, 0, stream>>>(segm, segmmean);
  k_style  <<<16, 256, 0, stream>>>(style, scw, scb, segmmean, styleterm, stylemean);
  k_demod  <<<16, 256, 0, stream>>>(weight, stylemean, demod);
  k_wcvt   <<<(9 * C_OUT * C_IN + 255) / 256, 256, 0, stream>>>(weight, w16);
  k_xmod   <<<dim3(NPP, 2), 256, 0, stream>>>(input, scw, segm, styleterm, xpad);
  k_conv_wmma<<<dim3(32, 4, 8), 256, 0, stream>>>(xpad, w16, demod, out);
}
